// Seq2SQL_v1_26121991094395
// MI455X (gfx1250) — matrix operand order repaired and checked by reading, not compiled, not run on hardware
//
#include <hip/hip_runtime.h>
#include <math.h>

// ---- CDNA5 WMMA types ------------------------------------------------------
typedef __bf16  v16bf __attribute__((ext_vector_type(16)));
typedef __bf16  v8bf  __attribute__((ext_vector_type(8)));
typedef __bf16  v4bf  __attribute__((ext_vector_type(4)));
typedef float   v8f   __attribute__((ext_vector_type(8)));

union AFrag { v16bf v; v8bf h[2]; };

// ---------------------------------------------------------------------------
// WMMA GEMM with 1x4 register blocking:
//   C[M,N](f32) = A[M,K](bf16) * B[N,K](bf16)^T + biasN[n] + rowBias[row/rowDiv,n]
// One wave computes a 16x64 strip (4 accumulators); the A fragment is loaded
// once per K-step and reused for 4 v_wmma_f32_16x16x32_bf16 issues, cutting
// L2 operand traffic from 2KB/WMMA to 1.25KB/WMMA and A re-fetch by 4x.
// Fragment layouts per CDNA5 ISA 7.12.2 (16-bit A 16x32, B 32x16, f32 C 16x16).
// ---------------------------------------------------------------------------
__global__ void wmma_gemm_bf16(const __bf16* __restrict__ A,
                               const __bf16* __restrict__ Bm,
                               float* __restrict__ C,
                               const float* __restrict__ biasN,
                               const float* __restrict__ rowBias, int rowDiv,
                               int Ndim, int Kdim,
                               int tilesN4, int tilesTotal) {
  const int lane = threadIdx.x & 31;
  const int wave = threadIdx.x >> 5;
  const int tile = blockIdx.x * 4 + wave;
  if (tile >= tilesTotal) return;            // whole-wave uniform

  const int hf = lane >> 4;                  // lane-group half (0/1)
  const int lr = lane & 15;                  // M (A rows) / N (B rows, C cols)
  const int tm = tile / tilesN4;
  const int tn = tile - tm * tilesN4;
  const int m0 = tm << 4;
  const int n0 = tn << 6;                    // 64-wide strip

  const __bf16* __restrict__ Arow = A + (size_t)(m0 + lr) * Kdim;
  const __bf16* __restrict__ Brow0 = Bm + (size_t)(n0 + lr) * Kdim;
  const size_t bstride = (size_t)16 * Kdim;  // next 16-col tile of B

  v8f acc0 = {}, acc1 = {}, acc2 = {}, acc3 = {};
  for (int k0 = 0; k0 < Kdim; k0 += 32) {
    AFrag a;
    // A lane: row (m0+lr), K = 16*hf..16*hf+15 (contiguous)
    a.h[0] = *(const v8bf*)(Arow + k0 + 16 * hf);
    a.h[1] = *(const v8bf*)(Arow + k0 + 16 * hf + 8);
    // B lane: column (n0+j*16+lr), K = 16*hf..16*hf+15 (contiguous 32B)
    const int kb = k0 + 16 * hf;
    v16bf b0 = *(const v16bf*)(Brow0 + kb);
    v16bf b1 = *(const v16bf*)(Brow0 + bstride + kb);
    v16bf b2 = *(const v16bf*)(Brow0 + 2 * bstride + kb);
    v16bf b3 = *(const v16bf*)(Brow0 + 3 * bstride + kb);
    acc0 = __builtin_amdgcn_wmma_f32_16x16x32_bf16(false, a.v, false, b0, (short)0, acc0, false, false);
    acc1 = __builtin_amdgcn_wmma_f32_16x16x32_bf16(false, a.v, false, b1, (short)0, acc1, false, false);
    acc2 = __builtin_amdgcn_wmma_f32_16x16x32_bf16(false, a.v, false, b2, (short)0, acc2, false, false);
    acc3 = __builtin_amdgcn_wmma_f32_16x16x32_bf16(false, a.v, false, b3, (short)0, acc3, false, false);
  }

  // C/D: lane -> N = n0 + j*16 + lr ; VGPR r -> M = m0 + (hf?8:0) + r
  const int mb = m0 + (hf ? 8 : 0);
  v8f accs[4] = {acc0, acc1, acc2, acc3};
#pragma unroll
  for (int j = 0; j < 4; ++j) {
    const int n = n0 + j * 16 + lr;
    const float bN = biasN ? biasN[n] : 0.0f;
#pragma unroll
    for (int r = 0; r < 8; ++r) {
      const int row = mb + r;
      float v = accs[j][r] + bN;
      if (rowBias) v += rowBias[(size_t)(row / rowDiv) * Ndim + n];
      C[(size_t)row * Ndim + n] = v;
    }
  }
}

// ---- f32 -> bf16 converters (float4 vectorized; all sizes are /4) ----------
__global__ void cvt_bf16_kernel(const float* __restrict__ in,
                                __bf16* __restrict__ out, int n4) {
  int i = blockIdx.x * 256 + threadIdx.x;
  if (i < n4) {
    float4 v = ((const float4*)in)[i];
    v4bf o = {(__bf16)v.x, (__bf16)v.y, (__bf16)v.z, (__bf16)v.w};
    *(v4bf*)(out + 4 * (size_t)i) = o;
  }
}

// out[r*cols+c] = in[r*inStride + inOff + c]   (extract enc_w halves); cols%4==0
__global__ void cvt_slice_kernel(const float* __restrict__ in,
                                 __bf16* __restrict__ out,
                                 int rows, int cols, int inStride, int inOff) {
  int i = blockIdx.x * 256 + threadIdx.x;          // index over groups of 4
  int total4 = rows * (cols >> 2);
  if (i < total4) {
    int c4pr = cols >> 2;
    int r = i / c4pr, c = (i - r * c4pr) << 2;
    float4 v = *(const float4*)(in + (size_t)r * inStride + inOff + c);
    v4bf o = {(__bf16)v.x, (__bf16)v.y, (__bf16)v.z, (__bf16)v.w};
    *(v4bf*)(out + (size_t)r * cols + c) = o;
  }
}

// out[j*n+i] = in[i*n+j]  (square), LDS-tiled so both sides coalesce
__global__ void transpose_cvt_kernel(const float* __restrict__ in,
                                     __bf16* __restrict__ out, int n) {
  __shared__ float tile[16][17];
  int j = blockIdx.x * 16 + threadIdx.x;
  int i = blockIdx.y * 16 + threadIdx.y;
  tile[threadIdx.y][threadIdx.x] = in[(size_t)i * n + j];
  __syncthreads();
  int jo = blockIdx.x * 16 + threadIdx.y;   // output row = j
  int io = blockIdx.y * 16 + threadIdx.x;   // output col = i
  out[(size_t)jo * n + io] = (__bf16)tile[threadIdx.x][threadIdx.y];
}

// ---- cb[b] = dot(wenc_u[b,:], V_b) ----------------------------------------
__global__ void dotv_kernel(const float* __restrict__ U,
                            const float* __restrict__ v,
                            float* __restrict__ out, int K) {
  __shared__ float r[256];
  int b = blockIdx.x, tid = threadIdx.x;
  float s = 0.f;
  for (int i = tid; i < K; i += 256) s += U[(size_t)b * K + i] * v[i];
  r[tid] = s; __syncthreads();
  for (int o = 128; o > 0; o >>= 1) {
    if (tid < o) r[tid] += r[tid + o];
    __syncthreads();
  }
  if (tid == 0) out[b] = r[0];
}

// ---------------------------------------------------------------------------
// Fused attention + softmax + pooling, one workgroup (256 thr) per header t.
// Stages wemb[t] (16x1024) in LDS as bf16 (32KB) via float4 loads, so the
// 84MB wemb tensor crosses HBM exactly once (the pipeline's roofline term).
// attn[m] = dot(wemb[t,m,:], uV[b,:]) + cb[b]; p = softmax; pooled = p . wemb.
// pooled is emitted directly in bf16 (feeds the final WMMA GEMM).
// ---------------------------------------------------------------------------
__global__ void attn_pool_kernel(const float* __restrict__ wemb,
                                 const float* __restrict__ uV,
                                 const float* __restrict__ cb,
                                 __bf16* __restrict__ pooled, int L) {
  __shared__ __bf16 sW[16 * 1024];
  __shared__ float  sU[1024];
  __shared__ float  red[16][17];
  __shared__ float  sp[16];
  const int t = blockIdx.x;
  const int b = t / L;
  const int tid = threadIdx.x;

  const float4* __restrict__ w4 = (const float4*)(wemb + (size_t)t * 16 * 1024);
  for (int i = tid; i < (16 * 1024) / 4; i += 256) {
    float4 v = w4[i];
    v4bf o = {(__bf16)v.x, (__bf16)v.y, (__bf16)v.z, (__bf16)v.w};
    *(v4bf*)(sW + 4 * i) = o;
  }
  const float4* __restrict__ u4 = (const float4*)(uV + (size_t)b * 1024);
  for (int i = tid; i < 1024 / 4; i += 256)
    *(float4*)(sU + 4 * i) = u4[i];
  __syncthreads();

  const int m = tid >> 4, c = tid & 15;      // 16 tokens x 16 partial lanes
  float acc = 0.f;
  for (int i = c; i < 1024; i += 16) acc += (float)sW[m * 1024 + i] * sU[i];
  red[m][c] = acc;
  __syncthreads();

  if (tid < 16) {
    float s = 0.f;
    for (int j = 0; j < 16; ++j) s += red[tid][j];
    red[tid][16] = s + cb[b];
  }
  __syncthreads();
  if (tid < 16) {
    float mx = -INFINITY;
    for (int j = 0; j < 16; ++j) mx = fmaxf(mx, red[j][16]);
    float den = 0.f;
    for (int j = 0; j < 16; ++j) den += expf(red[j][16] - mx);
    sp[tid] = expf(red[tid][16] - mx) / den;
  }
  __syncthreads();

  for (int i = tid; i < 1024; i += 256) {
    float a = 0.f;
#pragma unroll
    for (int mm = 0; mm < 16; ++mm) a += sp[mm] * (float)sW[mm * 1024 + i];
    pooled[(size_t)t * 1024 + i] = (__bf16)a;
  }
}

// ---- TF-style LayerNorm over IS, out = gamma*x + beta (float4) -------------
__global__ void layernorm_kernel(const float* __restrict__ h,
                                 const float* __restrict__ gamma,
                                 const float* __restrict__ beta,
                                 float* __restrict__ out, int IS) {
  __shared__ float r1[256], r2[256];
  const int row = blockIdx.x, tid = threadIdx.x;
  const float4* __restrict__ x4 = (const float4*)(h + (size_t)row * IS);
  const int n4 = IS >> 2;
  float s = 0.f, s2 = 0.f;
  for (int i = tid; i < n4; i += 256) {
    float4 v = x4[i];
    s  += v.x + v.y + v.z + v.w;
    s2 += v.x * v.x + v.y * v.y + v.z * v.z + v.w * v.w;
  }
  r1[tid] = s; r2[tid] = s2; __syncthreads();
  for (int o = 128; o > 0; o >>= 1) {
    if (tid < o) { r1[tid] += r1[tid + o]; r2[tid] += r2[tid + o]; }
    __syncthreads();
  }
  const float mean = r1[0] / IS;
  const float var  = fmaxf(r2[0] / IS - mean * mean, 0.f);
  const float rstd = rsqrtf(var + 1e-12f);
  float4* __restrict__ o4 = (float4*)(out + (size_t)row * IS);
  const float4* __restrict__ g4 = (const float4*)gamma;
  const float4* __restrict__ b4 = (const float4*)beta;
  for (int i = tid; i < n4; i += 256) {
    float4 v = x4[i], g = g4[i], bb = b4[i];
    float4 r;
    r.x = g.x * ((v.x - mean) * rstd) + bb.x;
    r.y = g.y * ((v.y - mean) * rstd) + bb.y;
    r.z = g.z * ((v.z - mean) * rstd) + bb.z;
    r.w = g.w * ((v.w - mean) * rstd) + bb.w;
    o4[i] = r;
  }
}

// ---------------------------------------------------------------------------
extern "C" void kernel_launch(void* const* d_in, const int* in_sizes, int n_in,
                              void* d_out, int out_size, void* d_ws, size_t ws_size,
                              hipStream_t stream) {
  const float* ctx   = (const float*)d_in[0];
  const float* wemb  = (const float*)d_in[1];
  // d_in[2]=l_hpu, d_in[3]=l_hs: constant lengths (M,L) -> layout is exact reshape
  const float* Uw    = (const float*)d_in[4];
  const float* Ub    = (const float*)d_in[5];
  const float* Vw    = (const float*)d_in[6];
  const float* Vb    = (const float*)d_in[7];
  const float* Ew    = (const float*)d_in[8];
  const float* Eb    = (const float*)d_in[9];
  const float* gamma = (const float*)d_in[10];
  const float* beta  = (const float*)d_in[11];
  float* out = (float*)d_out;

  const int B  = in_sizes[3];                 // 64
  const int IS = in_sizes[0] / B;             // 1024
  const int T  = in_sizes[2];                 // 1280
  const int L  = T / B;                       // 20

  // -- workspace carve-out (256B aligned) --
  size_t off = 0;
  auto alloc = [&](size_t bytes) -> char* {
    char* p = (char*)d_ws + off;
    off = (off + bytes + 255) & ~(size_t)255;
    return p;
  };
  __bf16* ctx_bf   = (__bf16*)alloc((size_t)B * IS * 2);
  __bf16* Uw_bf    = (__bf16*)alloc((size_t)IS * IS * 2);
  __bf16* VwT_bf   = (__bf16*)alloc((size_t)IS * IS * 2);
  __bf16* encA_bf  = (__bf16*)alloc((size_t)IS * IS * 2);
  __bf16* encB_bf  = (__bf16*)alloc((size_t)IS * IS * 2);
  float*  wencu_f  = (float*)alloc((size_t)B * IS * 4);
  __bf16* wencu_bf = (__bf16*)alloc((size_t)B * IS * 2);
  float*  uV_f     = (float*)alloc((size_t)B * IS * 4);
  float*  cb_f     = (float*)alloc((size_t)B * 4);
  float*  ctxE_f   = (float*)alloc((size_t)B * IS * 4);
  __bf16* pooled_bf= (__bf16*)alloc((size_t)T * IS * 2);
  float*  h_f      = (float*)alloc((size_t)T * IS * 4);
  (void)ws_size; (void)n_in; (void)out_size;

  // -- 1) operand conversion to bf16 --
  {
    int n4 = (B * IS) / 4;
    cvt_bf16_kernel<<<(n4 + 255) / 256, 256, 0, stream>>>(ctx, ctx_bf, n4);
    n4 = (IS * IS) / 4;
    cvt_bf16_kernel<<<(n4 + 255) / 256, 256, 0, stream>>>(Uw, Uw_bf, n4);
    dim3 tg(IS / 16, IS / 16);
    transpose_cvt_kernel<<<tg, dim3(16, 16), 0, stream>>>(Vw, VwT_bf, IS);
    cvt_slice_kernel<<<(n4 + 255) / 256, 256, 0, stream>>>(Ew, encA_bf, IS, IS, 2 * IS, 0);
    cvt_slice_kernel<<<(n4 + 255) / 256, 256, 0, stream>>>(Ew, encB_bf, IS, IS, 2 * IS, IS);
  }

  const int tilesN4 = IS / 64;                // 16 strips of 64 columns

  // -- 2) wenc_u = ctx @ U_w^T + U_b      [B,IS] --
  {
    int tiles = (B / 16) * tilesN4;
    wmma_gemm_bf16<<<tiles / 4, 128, 0, stream>>>(
        ctx_bf, Uw_bf, wencu_f, Ub, nullptr, 1, IS, IS, tilesN4, tiles);
    int n4 = (B * IS) / 4;
    cvt_bf16_kernel<<<(n4 + 255) / 256, 256, 0, stream>>>(wencu_f, wencu_bf, n4);
    dotv_kernel<<<B, 256, 0, stream>>>(wencu_f, Vb, cb_f, IS);  // cb = u . V_b
  }

  // -- 3) uV = wenc_u @ V_w  (B = V_w^T, row-major [j,i])   [B,IS] --
  {
    int tiles = (B / 16) * tilesN4;
    wmma_gemm_bf16<<<tiles / 4, 128, 0, stream>>>(
        wencu_bf, VwT_bf, uV_f, nullptr, nullptr, 1, IS, IS, tilesN4, tiles);
  }

  // -- 4) ctxE = ctx @ enc_w[:,IS:]^T + enc_b   [B,IS] --
  {
    int tiles = (B / 16) * tilesN4;
    wmma_gemm_bf16<<<tiles / 4, 128, 0, stream>>>(
        ctx_bf, encB_bf, ctxE_f, Eb, nullptr, 1, IS, IS, tilesN4, tiles);
  }

  // -- 5) fused attention + softmax + pooling (single HBM pass over wemb) --
  attn_pool_kernel<<<T, 256, 0, stream>>>(wemb, uV_f, cb_f, pooled_bf, L);

  // -- 6) h = pooled @ enc_w[:,:IS]^T + ctxE[t/L]   [T,IS] --
  {
    int tiles = (T / 16) * tilesN4;            // 1280 waves
    wmma_gemm_bf16<<<tiles / 4, 128, 0, stream>>>(
        pooled_bf, encA_bf, h_f, nullptr, ctxE_f, L, IS, IS, tilesN4, tiles);
  }

  // -- 7) LayerNorm -> d_out  [B,L,IS] --
  layernorm_kernel<<<T, 256, 0, stream>>>(h_f, gamma, beta, out, IS);
}
